// ImprovedAttentionMechanism_79723182949002
// MI455X (gfx1250) — matrix operand order repaired and checked by reading, not compiled, not run on hardware
//
#include <hip/hip_runtime.h>
#include <hip/hip_fp16.h>

#define SEQ 2048
#define FEA 512
#define NH  8
#define HD  64

typedef _Float16 h16;
typedef __attribute__((ext_vector_type(16))) _Float16 v16h;
typedef __attribute__((ext_vector_type(8)))  _Float16 v8h;
typedef __attribute__((ext_vector_type(4)))  _Float16 v4h;
typedef __attribute__((ext_vector_type(8)))  float    v8f;

// ---- WMMA helpers (CDNA5 gfx1250, wave32) -------------------------------
__device__ __forceinline__ v8f wmma32(v16h a, v16h b, v8f c) {
    // D = A(16x32 f16) * B(32x16 f16) + C(16x16 f32)
    return __builtin_amdgcn_wmma_f32_16x16x32_f16(
        /*neg_a=*/false, a, /*neg_b=*/false, b,
        /*c_mod=*/(short)0, c, /*reuse_a=*/false, /*reuse_b=*/false);
}

// A-matrix 16x32 f16 layout (ISA 7.12.2): lane l (=lane&15), half (=lane>>4):
//   elems 0..7  = K = kb + 8*half + 0..7   (row M = l)
//   elems 8..15 = K = kb + 16 + 8*half + 0..7
__device__ __forceinline__ v16h load_a_frag(const h16* __restrict__ src,
                                            int row, int ld, int kb, int half) {
    const h16* p = src + row * ld + kb + half * 8;
    v8h c0 = *reinterpret_cast<const v8h*>(p);
    v8h c1 = *reinterpret_cast<const v8h*>(p + 16);
    v16h r;
#pragma unroll
    for (int i = 0; i < 8; ++i) { r[i] = c0[i]; r[i + 8] = c1[i]; }
    return r;
}

// B-matrix 32x16 f16 layout: lane l holds column n=l; elems 0..7 -> K=kb+8*half+0..7,
// elems 8..15 -> K=kb+16+8*half+0..7 (same as A). Source is stored
// N-major ([n][k], k contiguous) so this is two 16-byte loads.
__device__ __forceinline__ v16h load_b_frag(const h16* __restrict__ src,
                                            int nrow, int ld, int kb, int half) {
    const h16* p = src + nrow * ld + kb + half * 8;
    v8h c0 = *reinterpret_cast<const v8h*>(p);
    v8h c1 = *reinterpret_cast<const v8h*>(p + 16);
    v16h r;
#pragma unroll
    for (int i = 0; i < 8; ++i) { r[i] = c0[i]; r[i + 8] = c1[i]; }
    return r;
}

// Build A fragment from f32 LDS row (attn probabilities), converting to f16.
__device__ __forceinline__ v16h load_a_frag_lds(const float* __restrict__ sr,
                                                int j0, int half) {
    const float4* p = reinterpret_cast<const float4*>(sr + j0 + 8 * half);
    float4 f0 = p[0], f1 = p[1];   // K = j0+8h+0..7
    float4 f2 = p[4], f3 = p[5];   // K = j0+16+8h+0..7
    v16h a;
    a[0]=(h16)f0.x; a[1]=(h16)f0.y; a[2]=(h16)f0.z;  a[3]=(h16)f0.w;
    a[4]=(h16)f1.x; a[5]=(h16)f1.y; a[6]=(h16)f1.z;  a[7]=(h16)f1.w;
    a[8]=(h16)f2.x; a[9]=(h16)f2.y; a[10]=(h16)f2.z; a[11]=(h16)f2.w;
    a[12]=(h16)f3.x; a[13]=(h16)f3.y; a[14]=(h16)f3.z; a[15]=(h16)f3.w;
    return a;
}

// ---- Kernel 1: LayerNorm + positional-encoding row 0, f32 -> f16 --------
__global__ __launch_bounds__(256)
void ln_pe_kernel(const float* __restrict__ x, const float* __restrict__ g,
                  const float* __restrict__ b, h16* __restrict__ x16) {
    int wave = threadIdx.x >> 5, lane = threadIdx.x & 31;
    int row = blockIdx.x * 8 + wave;
    const float4* xr4 = reinterpret_cast<const float4*>(x + row * FEA);
    const float4* g4  = reinterpret_cast<const float4*>(g);
    const float4* b4  = reinterpret_cast<const float4*>(b);
    float4 v[4];
    float s = 0.f;
#pragma unroll
    for (int i = 0; i < 4; ++i) {
        v[i] = xr4[lane + i * 32];
        s += v[i].x + v[i].y + v[i].z + v[i].w;
    }
#pragma unroll
    for (int off = 16; off; off >>= 1) s += __shfl_xor(s, off, 32);
    float mu = s * (1.0f / FEA);
    float vs = 0.f;
#pragma unroll
    for (int i = 0; i < 4; ++i) {
        float dx = v[i].x - mu, dy = v[i].y - mu, dz = v[i].z - mu, dw = v[i].w - mu;
        vs += dx * dx + dy * dy + dz * dz + dw * dw;
    }
#pragma unroll
    for (int off = 16; off; off >>= 1) vs += __shfl_xor(vs, off, 32);
    float inv = rsqrtf(vs * (1.0f / FEA) + 1e-5f);
#pragma unroll
    for (int i = 0; i < 4; ++i) {
        int f4i = lane + i * 32;                 // float4 index; f = 4*f4i
        float4 gg = g4[f4i], bb = b4[f4i];
        v4h o;                                   // pe row0 = (0,1,0,1)
        o[0] = (h16)((v[i].x - mu) * inv * gg.x + bb.x);
        o[1] = (h16)((v[i].y - mu) * inv * gg.y + bb.y + 1.0f);
        o[2] = (h16)((v[i].z - mu) * inv * gg.z + bb.z);
        o[3] = (h16)((v[i].w - mu) * inv * gg.w + bb.w + 1.0f);
        *reinterpret_cast<v4h*>(x16 + row * FEA + 4 * f4i) = o;
    }
}

// ---- Kernel 2: W [h][f][d] f32 -> Wt [h][d][f] f16 (N-major for WMMA B) --
__global__ __launch_bounds__(256)
void wconv_kernel(const float* __restrict__ wq, const float* __restrict__ wk,
                  const float* __restrict__ wv, h16* __restrict__ wqt,
                  h16* __restrict__ wkt, h16* __restrict__ wvt) {
    int idx = blockIdx.x * 256 + threadIdx.x;          // over NH*FEA*HD
    int d = idx % HD, f = (idx / HD) % FEA, h = idx / (HD * FEA);
    int o = (h * HD + d) * FEA + f;
    wqt[o] = (h16)wq[idx];
    wkt[o] = (h16)wk[idx];
    wvt[o] = (h16)wv[idx];
}

// ---- Kernel 3: QKV projections via WMMA ---------------------------------
// One wave computes a 16x64 tile of one of {Q,K,V} for one head.
// 2-stage ping-pong: loads for one stage overlap WMMA of the other,
// with no register rotation copies.
__global__ __launch_bounds__(256)
void qkv_kernel(const h16* __restrict__ x16,
                const h16* __restrict__ wqt, const h16* __restrict__ wkt,
                const h16* __restrict__ wvt,
                h16* __restrict__ q16, h16* __restrict__ k16,
                h16* __restrict__ vt16) {
    int wid  = blockIdx.x * 8 + (threadIdx.x >> 5);    // 0..3071
    int lane = threadIdx.x & 31, l = lane & 15, half = lane >> 4;
    int m  = wid % 3;
    int h  = (wid / 3) % NH;
    int it = wid / (3 * NH);
    int i0 = it * 16;
    const h16* wt = (m == 0) ? wqt : (m == 1) ? wkt : wvt;
    const h16* wth = wt + h * HD * FEA;                // [d][f]
    v8f acc[4] = {};

    v16h aA = load_a_frag(x16, i0 + l, FEA, 0, half);
    v16h bA[4];
#pragma unroll
    for (int nt = 0; nt < 4; ++nt) bA[nt] = load_b_frag(wth, nt * 16 + l, FEA, 0, half);

    for (int kb = 0; kb < FEA - 64; kb += 64) {        // 7 ping-pong iterations
        v16h aB = load_a_frag(x16, i0 + l, FEA, kb + 32, half);
        v16h bB[4];
#pragma unroll
        for (int nt = 0; nt < 4; ++nt)
            bB[nt] = load_b_frag(wth, nt * 16 + l, FEA, kb + 32, half);
#pragma unroll
        for (int nt = 0; nt < 4; ++nt) acc[nt] = wmma32(aA, bA[nt], acc[nt]);

        aA = load_a_frag(x16, i0 + l, FEA, kb + 64, half);
#pragma unroll
        for (int nt = 0; nt < 4; ++nt)
            bA[nt] = load_b_frag(wth, nt * 16 + l, FEA, kb + 64, half);
#pragma unroll
        for (int nt = 0; nt < 4; ++nt) acc[nt] = wmma32(aB, bB[nt], acc[nt]);
    }
    {   // tail: kb = FEA-64 and FEA-32
        v16h aB = load_a_frag(x16, i0 + l, FEA, FEA - 32, half);
        v16h bB[4];
#pragma unroll
        for (int nt = 0; nt < 4; ++nt)
            bB[nt] = load_b_frag(wth, nt * 16 + l, FEA, FEA - 32, half);
#pragma unroll
        for (int nt = 0; nt < 4; ++nt) acc[nt] = wmma32(aA, bA[nt], acc[nt]);
#pragma unroll
        for (int nt = 0; nt < 4; ++nt) acc[nt] = wmma32(aB, bB[nt], acc[nt]);
    }

    if (m == 2) {                                      // V: transposed store
        h16* vt = vt16 + h * HD * SEQ;
#pragma unroll
        for (int nt = 0; nt < 4; ++nt)
#pragma unroll
            for (int r = 0; r < 8; ++r)
                vt[(nt * 16 + l) * SEQ + i0 + r + 8 * half] = (h16)acc[nt][r];
    } else {
        h16* out = ((m == 0) ? q16 : k16) + (h * SEQ + i0) * HD;
#pragma unroll
        for (int nt = 0; nt < 4; ++nt)
#pragma unroll
            for (int r = 0; r < 8; ++r)
                out[(r + 8 * half) * HD + nt * 16 + l] = (h16)acc[nt][r];
    }
}

// One 16x16 score tile: 2 WMMAs + bias/scale epilogue into LDS.
__device__ __forceinline__ void score_tile(float* __restrict__ s_scores,
                                           v16h a0, v16h a1, v16h b0, v16h b1,
                                           int i0, int j0, int l, int half) {
    v8f acc = {};
    acc = wmma32(a0, b0, acc);
    acc = wmma32(a1, b1, acc);
#pragma unroll
    for (int r = 0; r < 8; ++r) {
        int rl = r + 8 * half;
        float diff = (float)((i0 + rl) - (j0 + l));
        s_scores[rl * SEQ + j0 + l] = acc[r] * 0.125f + __expf(-0.5f * diff * diff);
    }
}

// ---- Kernel 4: fused scores + bias + softmax + mean + attn*V ------------
// One block = 16 query rows, loops all 8 heads. LDS holds the full 16x2048
// score rows (128KB), the head-mean accumulator (128KB) and the 16x64 AV
// tile (4KB) -- 260KB of the 320KB WGP LDS.
__global__ __launch_bounds__(256)
void attn_kernel(const h16* __restrict__ q16, const h16* __restrict__ k16,
                 const h16* __restrict__ vt16, float* __restrict__ out) {
    extern __shared__ float smem[];
    float* s_scores = smem;                  // [16][SEQ]
    float* s_mean   = smem + 16 * SEQ;       // [16][SEQ]
    float* s_out    = smem + 32 * SEQ;       // [16][HD]
    int tid = threadIdx.x;
    int wave = tid >> 5, lane = tid & 31, l = lane & 15, half = lane >> 4;
    int i0 = blockIdx.x * 16;
    float* mh_out   = out;                   // [SEQ][NH*HD]
    float* mean_out = out + SEQ * FEA;       // [SEQ][SEQ]

    for (int h = 0; h < NH; ++h) {
        const h16* qh = q16 + h * SEQ * HD;
        const h16* kh = k16 + h * SEQ * HD;
        const h16* vh = vt16 + h * HD * SEQ;

        // hint the next head's K/V streams into cache while we compute
        if (h + 1 < NH) {
            __builtin_prefetch(kh + SEQ * HD + wave * (SEQ * HD / 8), 0, 0);
            __builtin_prefetch(vh + HD * SEQ + wave * (HD * SEQ / 8), 0, 0);
        }

        // ---- scores = QK^T/8 + exp(-0.5*(i-j)^2), 16 j-tiles per wave ----
        // 2-stage ping-pong over j-tiles, no rotation copies.
        v16h a0 = load_a_frag(qh, i0 + l, HD, 0, half);
        v16h a1 = load_a_frag(qh, i0 + l, HD, 32, half);
        {
            int jb = wave * 256;                       // this wave's j range
            v16h b0A = load_b_frag(kh, jb + l, HD, 0,  half);
            v16h b1A = load_b_frag(kh, jb + l, HD, 32, half);
            for (int t = 0; t < 14; t += 2) {
                int j0 = jb + t * 16;
                v16h b0B = load_b_frag(kh, j0 + 16 + l, HD, 0,  half);
                v16h b1B = load_b_frag(kh, j0 + 16 + l, HD, 32, half);
                score_tile(s_scores, a0, a1, b0A, b1A, i0, j0, l, half);
                b0A = load_b_frag(kh, j0 + 32 + l, HD, 0,  half);
                b1A = load_b_frag(kh, j0 + 32 + l, HD, 32, half);
                score_tile(s_scores, a0, a1, b0B, b1B, i0, j0 + 16, l, half);
            }
            {   // tail: tiles 14, 15
                int j0 = jb + 14 * 16;
                v16h b0B = load_b_frag(kh, j0 + 16 + l, HD, 0,  half);
                v16h b1B = load_b_frag(kh, j0 + 16 + l, HD, 32, half);
                score_tile(s_scores, a0, a1, b0A, b1A, i0, j0, l, half);
                score_tile(s_scores, a0, a1, b0B, b1B, i0, j0 + 16, l, half);
            }
        }
        __syncthreads();

        // ---- softmax: each wave owns 2 full rows (float4 sweeps) ----
#pragma unroll
        for (int rr = 0; rr < 2; ++rr) {
            float4* sr = reinterpret_cast<float4*>(s_scores + (wave * 2 + rr) * SEQ);
            float mx = -1e30f;
            for (int c = lane; c < SEQ / 4; c += 32) {
                float4 q = sr[c];
                mx = fmaxf(mx, fmaxf(fmaxf(q.x, q.y), fmaxf(q.z, q.w)));
            }
#pragma unroll
            for (int off = 16; off; off >>= 1) mx = fmaxf(mx, __shfl_xor(mx, off, 32));
            float sum = 0.f;
            for (int c = lane; c < SEQ / 4; c += 32) {
                float4 q = sr[c];
                q.x = __expf(q.x - mx); q.y = __expf(q.y - mx);
                q.z = __expf(q.z - mx); q.w = __expf(q.w - mx);
                sr[c] = q;
                sum += q.x + q.y + q.z + q.w;
            }
#pragma unroll
            for (int off = 16; off; off >>= 1) sum += __shfl_xor(sum, off, 32);
            float inv = 1.0f / sum;
            for (int c = lane; c < SEQ / 4; c += 32) {
                float4 q = sr[c];
                q.x *= inv; q.y *= inv; q.z *= inv; q.w *= inv;
                sr[c] = q;
            }
        }
        __syncthreads();

        // ---- mean-over-heads accumulate + clear AV tile (float4) ----
        {
            float4* sm = reinterpret_cast<float4*>(s_mean);
            const float4* ss = reinterpret_cast<const float4*>(s_scores);
            if (h == 0)
                for (int idx = tid; idx < 16 * SEQ / 4; idx += 256) {
                    float4 q = ss[idx];
                    sm[idx] = make_float4(q.x * 0.125f, q.y * 0.125f,
                                          q.z * 0.125f, q.w * 0.125f);
                }
            else
                for (int idx = tid; idx < 16 * SEQ / 4; idx += 256) {
                    float4 q = ss[idx], mo = sm[idx];
                    sm[idx] = make_float4(mo.x + q.x * 0.125f, mo.y + q.y * 0.125f,
                                          mo.z + q.z * 0.125f, mo.w + q.w * 0.125f);
                }
            float4* so = reinterpret_cast<float4*>(s_out);
            if (tid < 16 * HD / 4) so[tid] = make_float4(0.f, 0.f, 0.f, 0.f);
        }
        __syncthreads();

        // ---- attn * V: waves split the K(j) dimension, ds_add_f32 reduce --
        v8f acc[4] = {};
        {
            int ks0 = wave * 8;
            const float* srow = s_scores + l * SEQ;
            v16h bA[4];
#pragma unroll
            for (int nt = 0; nt < 4; ++nt)
                bA[nt] = load_b_frag(vh, nt * 16 + l, SEQ, ks0 * 32, half);
            for (int s = 0; s < 6; s += 2) {           // 3 ping-pong iterations
                int j0 = (ks0 + s) * 32;
                v16h bB[4];
#pragma unroll
                for (int nt = 0; nt < 4; ++nt)
                    bB[nt] = load_b_frag(vh, nt * 16 + l, SEQ, j0 + 32, half);
                v16h a = load_a_frag_lds(srow, j0, half);
#pragma unroll
                for (int nt = 0; nt < 4; ++nt) acc[nt] = wmma32(a, bA[nt], acc[nt]);
#pragma unroll
                for (int nt = 0; nt < 4; ++nt)
                    bA[nt] = load_b_frag(vh, nt * 16 + l, SEQ, j0 + 64, half);
                a = load_a_frag_lds(srow, j0 + 32, half);
#pragma unroll
                for (int nt = 0; nt < 4; ++nt) acc[nt] = wmma32(a, bB[nt], acc[nt]);
            }
            {   // tail: s = 6, 7
                int j0 = (ks0 + 6) * 32;
                v16h bB[4];
#pragma unroll
                for (int nt = 0; nt < 4; ++nt)
                    bB[nt] = load_b_frag(vh, nt * 16 + l, SEQ, j0 + 32, half);
                v16h a = load_a_frag_lds(srow, j0, half);
#pragma unroll
                for (int nt = 0; nt < 4; ++nt) acc[nt] = wmma32(a, bA[nt], acc[nt]);
                a = load_a_frag_lds(srow, j0 + 32, half);
#pragma unroll
                for (int nt = 0; nt < 4; ++nt) acc[nt] = wmma32(a, bB[nt], acc[nt]);
            }
        }
#pragma unroll
        for (int nt = 0; nt < 4; ++nt)
#pragma unroll
            for (int r = 0; r < 8; ++r)
                atomicAdd(&s_out[(r + 8 * half) * HD + nt * 16 + l], acc[nt][r]);
        __syncthreads();

        {                                              // 16x64 tile, float4
            int row = tid >> 4, colv = tid & 15;
            reinterpret_cast<float4*>(mh_out + (i0 + row) * FEA + h * HD)[colv] =
                reinterpret_cast<const float4*>(s_out + row * HD)[colv];
        }
        __syncthreads();
    }

    {
        const float4* sm = reinterpret_cast<const float4*>(s_mean);
        for (int idx = tid; idx < 16 * SEQ / 4; idx += 256) {
            int row = idx >> 9, jv = idx & 511;        // SEQ/4 = 512 per row
            reinterpret_cast<float4*>(mean_out + (i0 + row) * SEQ)[jv] = sm[idx];
        }
    }
}

// ---- host launcher -------------------------------------------------------
extern "C" void kernel_launch(void* const* d_in, const int* in_sizes, int n_in,
                              void* d_out, int out_size, void* d_ws, size_t ws_size,
                              hipStream_t stream) {
    const float* x  = (const float*)d_in[0];
    const float* wq = (const float*)d_in[1];
    const float* wk = (const float*)d_in[2];
    const float* wv = (const float*)d_in[3];
    const float* lg = (const float*)d_in[4];
    const float* lb = (const float*)d_in[5];

    char* ws = (char*)d_ws;
    h16* X16  = (h16*)(ws + 0);           // 2048*512*2      = 2,097,152
    h16* WQT  = (h16*)(ws + 2097152);     // 8*512*64*2      =   524,288
    h16* WKT  = (h16*)(ws + 2621440);
    h16* WVT  = (h16*)(ws + 3145728);
    h16* Q16  = (h16*)(ws + 3670016);     // 8*2048*64*2     = 2,097,152
    h16* K16  = (h16*)(ws + 5767168);
    h16* VT16 = (h16*)(ws + 7864320);     // [h][64][2048]

    ln_pe_kernel<<<SEQ / 8, 256, 0, stream>>>(x, lg, lb, X16);
    wconv_kernel<<<(NH * FEA * HD) / 256, 256, 0, stream>>>(wq, wk, wv, WQT, WKT, WVT);
    qkv_kernel<<<(SEQ / 16) * NH * 3 / 8, 256, 0, stream>>>(X16, WQT, WKT, WVT,
                                                            Q16, K16, VT16);

    const int smem_bytes = (32 * SEQ + 16 * HD) * (int)sizeof(float); // 266,240 B
    (void)hipFuncSetAttribute((const void*)attn_kernel,
                              hipFuncAttributeMaxDynamicSharedMemorySize, smem_bytes);
    attn_kernel<<<SEQ / 16, 256, smem_bytes, stream>>>(Q16, K16, VT16, (float*)d_out);
}
